// GCNMassSpecPredictor_23141283791390
// MI455X (gfx1250) — matrix operand order repaired and checked by reading, not compiled, not run on hardware
//
#include <hip/hip_runtime.h>
#include <hip/hip_bf16.h>
#include <cfloat>

typedef __attribute__((ext_vector_type(16))) _Float16 v16h;
typedef __attribute__((ext_vector_type(8)))  _Float16 h8;
typedef __attribute__((ext_vector_type(4)))  _Float16 h4;
typedef __attribute__((ext_vector_type(2)))  _Float16 h2;
typedef __attribute__((ext_vector_type(8)))  float    v8f;

#define EPS_BN 1e-5f
#define TILE_M 32
#define TILE_N 128
#define TILE_K 32
#define LDS_STRIDE 40   // padded f16 row stride (80B): bank-conflict-free b128 frag loads

// ---------------------------------------------------------------------------
// f32 -> f16 weight conversion (once per call; weights are tiny vs activations)
// ---------------------------------------------------------------------------
__global__ void k_cvt_f16(const float* __restrict__ in, _Float16* __restrict__ out, size_t n) {
  size_t i = (size_t)blockIdx.x * blockDim.x + threadIdx.x;
  for (; i < n; i += (size_t)gridDim.x * blockDim.x) out[i] = (_Float16)in[i];
}

// ---------------------------------------------------------------------------
// WMMA GEMM: C[M,Nc] = A[M,K](f32) @ B[K,Nc](f16) (+bias) (+ReLU)
// 256 threads = 8 waves; block tile 32x128; wave tile 16x32 (2 WMMA, shared A);
// K step 32. Double-buffered LDS, register-staged pipeline, 1 sync per step.
// ---------------------------------------------------------------------------
struct PanelRegs { h4 a; h2 b[8]; };

__device__ __forceinline__ PanelRegs load_panel(
    const float* __restrict__ A, int lda, int rowBase, int K,
    const _Float16* __restrict__ Bw, int ldb, int colBase, int Nc,
    int k0, int tid)
{
  PanelRegs pr;
  // ---- A: 32 rows x 32 k, 4 consecutive k per thread ----
  {
    const int r  = tid >> 3;
    const int kk = (tid & 7) * 4;
    const size_t base = (size_t)(rowBase + r) * lda + k0 + kk;
    if (k0 + TILE_K <= K) {                       // fast: unconditional float4
      const float4 v = *(const float4*)(A + base);
      pr.a = (h4){(_Float16)v.x, (_Float16)v.y, (_Float16)v.z, (_Float16)v.w};
    } else {                                      // tail: branch-free clamp+select
#pragma unroll
      for (int i = 0; i < 4; ++i) {
        const bool ok = (k0 + kk + i) < K;
        const float v = A[ok ? (base + i) : (size_t)0];
        pr.a[i] = (_Float16)(ok ? v : 0.0f);
      }
    }
  }
  // ---- B: 32 k x 128 n, thread owns k-pair kp and 8 n ----
  {
    const int kp = tid >> 4;            // 0..15
    const int n0 = (tid & 15) * 8;
    const int gk = k0 + 2 * kp;
    if (k0 + TILE_K <= K && colBase + TILE_N <= Nc) {   // fast: 2x b128 f16 loads
      const h8 r0 = *(const h8*)(Bw + (size_t)gk * ldb + colBase + n0);
      const h8 r1 = *(const h8*)(Bw + (size_t)(gk + 1) * ldb + colBase + n0);
#pragma unroll
      for (int i = 0; i < 8; ++i) { pr.b[i][0] = r0[i]; pr.b[i][1] = r1[i]; }
    } else {                                            // tail: branch-free
#pragma unroll
      for (int i = 0; i < 8; ++i) {
#pragma unroll
        for (int par = 0; par < 2; ++par) {
          const int kidx = gk + par;
          const int nidx = colBase + n0 + i;
          const bool ok = (kidx < K) && (nidx < Nc);
          const _Float16 v = Bw[ok ? ((size_t)kidx * ldb + nidx) : (size_t)0];
          pr.b[i][par] = ok ? v : (_Float16)0.0f;
        }
      }
    }
  }
  return pr;
}

__device__ __forceinline__ void store_panel(const PanelRegs& pr,
                                            _Float16* __restrict__ As,
                                            _Float16* __restrict__ Bs, int tid)
{
  { // A row-major [r][k], padded stride
    const int r  = tid >> 3;
    const int kk = (tid & 7) * 4;
    *(h4*)(As + r * LDS_STRIDE + kk) = pr.a;           // ds_store_b64
  }
  { // B swizzled [n][kp][par], padded stride
    const int kp = tid >> 4;
    const int n0 = (tid & 15) * 8;
#pragma unroll
    for (int i = 0; i < 8; ++i)
      *(h2*)(Bs + (n0 + i) * LDS_STRIDE + kp * 2) = pr.b[i];  // merged ds_store_2addr_b32
  }
}

__global__ __launch_bounds__(256) void gemm_wmma_f16(
    const float* __restrict__ A, int lda,
    const _Float16* __restrict__ Bw, int ldb,
    const float* __restrict__ bias,
    float* __restrict__ C, int ldc,
    int M, int K, int Nc, int fuseRelu)
{
  __shared__ _Float16 AsBuf[2][TILE_M * LDS_STRIDE];
  __shared__ _Float16 BsBuf[2][TILE_N * LDS_STRIDE];

  const int tid  = threadIdx.x;
  const int wave = tid >> 5;
  const int lane = tid & 31;
  const int wr   = wave >> 2;     // 0..1  (16-row subtile)
  const int wc   = wave & 3;      // 0..3  (32-col subtile)
  const int half = lane >> 4;
  const int lsub = lane & 15;

  const int rowBase = blockIdx.x * TILE_M;
  const int colBase = blockIdx.y * TILE_N;
  const int nSteps  = (K + TILE_K - 1) / TILE_K;

  v8f acc0 = {}, acc1 = {};

  PanelRegs pr = load_panel(A, lda, rowBase, K, Bw, ldb, colBase, Nc, 0, tid);
  store_panel(pr, AsBuf[0], BsBuf[0], tid);
  __syncthreads();

  for (int s = 0; s < nSteps; ++s) {
    const int cur = s & 1;

    if (s + 1 < nSteps)   // issue next panel's global loads before the math
      pr = load_panel(A, lda, rowBase, K, Bw, ldb, colBase, Nc, (s + 1) * TILE_K, tid);
    if (s + 2 < nSteps) { // WGP-scope global_prefetch_b8 on panel s+2
      __builtin_prefetch(&A[(size_t)(rowBase + (tid >> 3)) * lda + (s + 2) * TILE_K], 0, 3);
      __builtin_prefetch(&Bw[(size_t)((s + 2) * TILE_K + (tid >> 4)) * ldb + colBase], 0, 3);
    }

    // ---- fragments: A = two contiguous 8xf16 K-runs; B pre-swizzled ----
    const _Float16* As = AsBuf[cur];
    const _Float16* Bs = BsBuf[cur];
    v16h a, b0, b1;
    {
      const int arow = (wr * 16 + lsub) * LDS_STRIDE;
      const h8 lo = *(const h8*)(As + arow + 16 * half);       // ds_load_b128
      const h8 hi = *(const h8*)(As + arow + 16 * half + 8);   // ds_load_b128
      a = __builtin_shufflevector(lo, hi, 0,1,2,3,4,5,6,7,8,9,10,11,12,13,14,15);
    }
    {
      const int brow = (wc * 32 + lsub) * LDS_STRIDE + 16 * half;
      const h8 lo = *(const h8*)(Bs + brow);                   // ds_load_b128
      const h8 hi = *(const h8*)(Bs + brow + 8);               // ds_load_b128
      b0 = __builtin_shufflevector(lo, hi, 0,1,2,3,4,5,6,7,8,9,10,11,12,13,14,15);
    }
    {
      const int brow = (wc * 32 + 16 + lsub) * LDS_STRIDE + 16 * half;
      const h8 lo = *(const h8*)(Bs + brow);
      const h8 hi = *(const h8*)(Bs + brow + 8);
      b1 = __builtin_shufflevector(lo, hi, 0,1,2,3,4,5,6,7,8,9,10,11,12,13,14,15);
    }

    acc0 = __builtin_amdgcn_wmma_f32_16x16x32_f16(false, a, false, b0, (short)0, acc0, false, false);
    acc1 = __builtin_amdgcn_wmma_f32_16x16x32_f16(false, a, false, b1, (short)0, acc1, false, false);

    if (s + 1 < nSteps) store_panel(pr, AsBuf[cur ^ 1], BsBuf[cur ^ 1], tid);
    __syncthreads();
  }

  // ---- epilogue: VGPR v holds row (wr*16 + 8*half + v), col lsub ----
#pragma unroll
  for (int sub = 0; sub < 2; ++sub) {
    const v8f acc = sub ? acc1 : acc0;
    const int col = colBase + wc * 32 + sub * 16 + lsub;
    if (col < Nc) {
      const float bv = bias ? bias[col] : 0.0f;
#pragma unroll
      for (int v = 0; v < 8; ++v) {
        const int row = rowBase + wr * 16 + 8 * half + v;
        float out = acc[v] + bv;
        if (fuseRelu) out = fmaxf(out, 0.0f);
        C[(size_t)row * ldc + col] = out;
      }
    }
  }
}

// ---------------------------------------------------------------------------
// Elementwise / reduction helpers
// ---------------------------------------------------------------------------
__global__ void k_zero(float* p, size_t n) {
  size_t i = (size_t)blockIdx.x * blockDim.x + threadIdx.x;
  for (; i < n; i += (size_t)gridDim.x * blockDim.x) p[i] = 0.0f;
}

__global__ void k_fill(float* p, float v, size_t n) {
  size_t i = (size_t)blockIdx.x * blockDim.x + threadIdx.x;
  for (; i < n; i += (size_t)gridDim.x * blockDim.x) p[i] = v;
}

__global__ void k_edge_deg(const int* __restrict__ dst, float* __restrict__ deg, int E) {
  int e = blockIdx.x * blockDim.x + threadIdx.x;
  if (e < E) atomicAdd(&deg[dst[e]], 1.0f);
}

__global__ void k_dinv(float* __restrict__ deg, int N) {
  int i = blockIdx.x * blockDim.x + threadIdx.x;
  if (i < N) deg[i] = rsqrtf(deg[i] + 1.0f);
}

// Coalesced column stats, stage 1: threads -> columns, blocks -> row chunks.
__global__ __launch_bounds__(256) void k_colstats_part(
    const float* __restrict__ X, int R, int C, int rowsPerBlock,
    float* __restrict__ sum, float* __restrict__ sq)
{
  const int r0 = blockIdx.x * rowsPerBlock;
  const int r1 = min(r0 + rowsPerBlock, R);
  for (int c = threadIdx.x; c < C; c += 256) {
    float s = 0.0f, q = 0.0f;
    for (int r = r0; r < r1; ++r) {
      float v = X[(size_t)r * C + c];
      s += v; q += v * v;
    }
    atomicAdd(&sum[c], s);
    atomicAdd(&sq[c], q);
  }
}

__global__ void k_colstats_fin(float* __restrict__ sum, float* __restrict__ sq,
                               int R, int C)
{
  int c = blockIdx.x * blockDim.x + threadIdx.x;
  if (c >= C) return;
  float m = sum[c] / (float)R;
  sum[c] = m;                        // sum -> mean
  sq[c]  = sq[c] / (float)R - m * m; // sq -> var
}

__global__ void k_bn_relu(float* __restrict__ X, int R, int C,
                          const float* __restrict__ mean, const float* __restrict__ var,
                          const float* __restrict__ g, const float* __restrict__ b)
{
  size_t i = (size_t)blockIdx.x * blockDim.x + threadIdx.x;
  if (i >= (size_t)R * C) return;
  int c = (int)(i % C);
  float v = (X[i] - mean[c]) * rsqrtf(var[c] + EPS_BN) * g[c] + b[c];
  X[i] = fmaxf(v, 0.0f);
}

__global__ void k_bn_relu_res(float* __restrict__ h, const float* __restrict__ agg,
                              int R, int C,
                              const float* __restrict__ mean, const float* __restrict__ var,
                              const float* __restrict__ g, const float* __restrict__ b)
{
  size_t i = (size_t)blockIdx.x * blockDim.x + threadIdx.x;
  if (i >= (size_t)R * C) return;
  int c = (int)(i % C);
  float v = (agg[i] - mean[c]) * rsqrtf(var[c] + EPS_BN) * g[c] + b[c];
  h[i] += fmaxf(v, 0.0f);
}

__global__ void k_edge_scatter(const int* __restrict__ src, const int* __restrict__ dst,
                               const float* __restrict__ dinv, const float* __restrict__ hw,
                               float* __restrict__ agg, int E, int H)
{
  const int chunks = H >> 2;
  int idx = blockIdx.x * blockDim.x + threadIdx.x;
  if (idx >= E * chunks) return;
  const int e = idx / chunks;
  const int f = (idx - e * chunks) << 2;
  const int s = src[e], d = dst[e];
  const float norm = dinv[s] * dinv[d];
  const float4 v = *(const float4*)(hw + (size_t)s * H + f);
  float* out = agg + (size_t)d * H + f;
  atomicAdd(out + 0, v.x * norm);
  atomicAdd(out + 1, v.y * norm);
  atomicAdd(out + 2, v.z * norm);
  atomicAdd(out + 3, v.w * norm);
}

__global__ void k_self_bias(const float* __restrict__ hw, const float* __restrict__ dinv,
                            const float* __restrict__ bias, float* __restrict__ agg,
                            int N, int H)
{
  size_t i = (size_t)blockIdx.x * blockDim.x + threadIdx.x;
  if (i >= (size_t)N * H) return;
  int n = (int)(i / H);
  int f = (int)(i % H);
  float sn = dinv[n] * dinv[n];
  agg[i] += hw[i] * sn + bias[f];
}

// one wave per node: coalesced reads + wave32 shuffle reduction
__global__ __launch_bounds__(256) void k_gate2(
    const float* __restrict__ gh, const float* __restrict__ w,
    const float* __restrict__ bptr, float* __restrict__ gate, int N, int K)
{
  const int wave = threadIdx.x >> 5;
  const int lane = threadIdx.x & 31;
  const int n = blockIdx.x * 8 + wave;
  if (n >= N) return;
  float s = 0.0f;
  for (int k = lane; k < K; k += 32) s += gh[(size_t)n * K + k] * w[k];
#pragma unroll
  for (int off = 16; off > 0; off >>= 1) s += __shfl_down(s, off, 32);
  if (lane == 0) gate[n] = s + bptr[0];
}

__device__ __forceinline__ void atomicMaxF(float* addr, float val) {
  unsigned int* ua = (unsigned int*)addr;
  unsigned int old = *ua;
  while (__uint_as_float(old) < val) {
    unsigned int assumed = old;
    old = atomicCAS(ua, assumed, __float_as_uint(val));
    if (old == assumed) break;
  }
}

__global__ void k_segmax(const float* __restrict__ gate, const int* __restrict__ batch,
                         float* __restrict__ gmax, int N)
{
  int n = blockIdx.x * blockDim.x + threadIdx.x;
  if (n < N) atomicMaxF(&gmax[batch[n]], gate[n]);
}

__global__ void k_expsum(const float* __restrict__ gate, const int* __restrict__ batch,
                         const float* __restrict__ gmax, float* __restrict__ e,
                         float* __restrict__ denom, int N)
{
  int n = blockIdx.x * blockDim.x + threadIdx.x;
  if (n >= N) return;
  int b = batch[n];
  float v = __expf(gate[n] - gmax[b]);
  e[n] = v;
  atomicAdd(&denom[b], v);
}

__global__ void k_pool(const float* __restrict__ t, const float* __restrict__ e,
                       const int* __restrict__ batch, float* __restrict__ pooled,
                       int N, int H)
{
  const int chunks = H >> 2;
  int idx = blockIdx.x * blockDim.x + threadIdx.x;
  if (idx >= N * chunks) return;
  const int n = idx / chunks;
  const int f = (idx - n * chunks) << 2;
  const int b = batch[n];
  const float w = e[n];
  const float4 v = *(const float4*)(t + (size_t)n * H + f);
  float* out = pooled + (size_t)b * H + f;
  atomicAdd(out + 0, v.x * w);
  atomicAdd(out + 1, v.y * w);
  atomicAdd(out + 2, v.z * w);
  atomicAdd(out + 3, v.w * w);
}

__global__ void k_pool_div(float* __restrict__ pooled, const float* __restrict__ denom,
                           int B, int H)
{
  size_t i = (size_t)blockIdx.x * blockDim.x + threadIdx.x;
  if (i >= (size_t)B * H) return;
  int b = (int)(i / H);
  pooled[i] /= (denom[b] + 1e-12f);
}

__global__ void k_sigmoid(float* __restrict__ X, size_t n) {
  size_t i = (size_t)blockIdx.x * blockDim.x + threadIdx.x;
  if (i < n) X[i] = 1.0f / (1.0f + __expf(-X[i]));
}

// ---------------------------------------------------------------------------
// Launcher
// ---------------------------------------------------------------------------
static inline unsigned ceil_div(size_t a, unsigned b) { return (unsigned)((a + b - 1) / b); }

extern "C" void kernel_launch(void* const* d_in, const int* in_sizes, int n_in,
                              void* d_out, int out_size, void* d_ws, size_t ws_size,
                              hipStream_t stream)
{
  const float* x     = (const float*)d_in[0];
  const int*   ei    = (const int*)  d_in[1];
  const int*   batch = (const int*)  d_in[2];
  const float* embW  = (const float*)d_in[3];
  const float* embB  = (const float*)d_in[4];
  const float* embG  = (const float*)d_in[5];
  const float* embBe = (const float*)d_in[6];
  const float* convW = (const float*)d_in[7];
  const float* convB = (const float*)d_in[8];
  const float* convG = (const float*)d_in[9];
  const float* convBe= (const float*)d_in[10];
  const float* gW1   = (const float*)d_in[11];
  const float* gB1   = (const float*)d_in[12];
  const float* gW2   = (const float*)d_in[13];
  const float* gB2   = (const float*)d_in[14];
  const float* tW    = (const float*)d_in[15];
  const float* tB    = (const float*)d_in[16];
  const float* mW1   = (const float*)d_in[17];
  const float* mB1   = (const float*)d_in[18];
  const float* mG1   = (const float*)d_in[19];
  const float* mBe1  = (const float*)d_in[20];
  const float* mW2   = (const float*)d_in[21];
  const float* mB2   = (const float*)d_in[22];
  const float* mG2   = (const float*)d_in[23];
  const float* mBe2  = (const float*)d_in[24];
  const float* mW3   = (const float*)d_in[25];
  const float* mB3   = (const float*)d_in[26];

  const int N  = in_sizes[2];            // 100000
  const int E  = in_sizes[1] / 2;        // 400000
  const int H  = in_sizes[4];            // 256
  const int F  = in_sizes[0] / N;        // 44
  const int L  = in_sizes[8] / H;        // 5
  const int Hg = in_sizes[12];           // 128
  const int H2 = in_sizes[18];           // 512
  const int S  = in_sizes[26];           // 1000
  const int B  = out_size / S;           // 2048

  const int* src = ei;
  const int* dst = ei + E;

  // ---- workspace carve-up ----
  float* ws = (float*)d_ws;
  const size_t NH = (size_t)N * H;
  float* h      = ws;
  float* buf1   = ws + NH;               // hw / gateH / h1
  float* buf2   = ws + 2 * NH;           // agg / t  / h2
  float* dinv   = ws + 3 * NH;
  float* gate   = dinv + N;
  float* eArr   = gate + N;
  float* gmax   = eArr + N;
  float* denom  = gmax + B;
  float* pooled = denom + B;
  float* meanv  = pooled + (size_t)B * H; // stage-1 sums -> means
  float* varv   = meanv + 1024;           // stage-1 sumsq -> vars
  uintptr_t wp = (uintptr_t)(varv + 1024);
  wp = (wp + 15) & ~(uintptr_t)15;
  _Float16* w16pool = (_Float16*)wp;
  _Float16* embW16  = w16pool;                          // F*H
  _Float16* convW16 = embW16 + (size_t)F * H;           // L*H*H
  _Float16* gW1_16  = convW16 + (size_t)L * H * H;      // H*Hg
  _Float16* tW16    = gW1_16 + (size_t)H * Hg;          // H*H
  _Float16* mW1_16  = tW16 + (size_t)H * H;             // H*H2
  _Float16* mW2_16  = mW1_16 + (size_t)H * H2;          // H2*H2
  _Float16* mW3_16  = mW2_16 + (size_t)H2 * H2;         // H2*S
  (void)ws_size; (void)n_in;

  const unsigned TB = 256;
  auto cvt = [&](const float* in, _Float16* out, size_t n) {
    hipLaunchKernelGGL(k_cvt_f16, dim3(ceil_div(n, TB)), dim3(TB), 0, stream, in, out, n);
  };
  auto gemm = [&](const float* A, int lda, const _Float16* W, int ldb, const float* bias,
                  float* C, int ldc, int M, int K, int Nc, int relu) {
    dim3 grid(M / TILE_M, (unsigned)((Nc + TILE_N - 1) / TILE_N));
    hipLaunchKernelGGL(gemm_wmma_f16, grid, dim3(TB), 0, stream,
                       A, lda, W, ldb, bias, C, ldc, M, K, Nc, relu);
  };
  auto colstats = [&](const float* X, int R, int C) {
    hipLaunchKernelGGL(k_zero, dim3(8), dim3(TB), 0, stream, meanv, (size_t)C);
    hipLaunchKernelGGL(k_zero, dim3(8), dim3(TB), 0, stream, varv, (size_t)C);
    const int rowsPerBlock = 256;
    hipLaunchKernelGGL(k_colstats_part, dim3(ceil_div(R, rowsPerBlock)), dim3(TB), 0, stream,
                       X, R, C, rowsPerBlock, meanv, varv);
    hipLaunchKernelGGL(k_colstats_fin, dim3(ceil_div(C, TB)), dim3(TB), 0, stream,
                       meanv, varv, R, C);
  };

  // 0) weights -> f16 (once per call)
  cvt(embW, embW16, (size_t)F * H);
  cvt(convW, convW16, (size_t)L * H * H);
  cvt(gW1, gW1_16, (size_t)H * Hg);
  cvt(tW, tW16, (size_t)H * H);
  cvt(mW1, mW1_16, (size_t)H * H2);
  cvt(mW2, mW2_16, (size_t)H2 * H2);
  cvt(mW3, mW3_16, (size_t)H2 * S);

  // 1) degrees -> dinv
  hipLaunchKernelGGL(k_zero, dim3(1024), dim3(TB), 0, stream, dinv, (size_t)N);
  hipLaunchKernelGGL(k_edge_deg, dim3(ceil_div(E, TB)), dim3(TB), 0, stream, dst, dinv, E);
  hipLaunchKernelGGL(k_dinv, dim3(ceil_div(N, TB)), dim3(TB), 0, stream, dinv, N);

  // 2) embedding: h = relu(BN(x @ embW + embB))
  gemm(x, F, embW16, H, embB, h, H, N, F, H, 0);
  colstats(h, N, H);
  hipLaunchKernelGGL(k_bn_relu, dim3(ceil_div(NH, TB)), dim3(TB), 0, stream,
                     h, N, H, meanv, varv, embG, embBe);

  // 3) GCN layers
  for (int l = 0; l < L; ++l) {
    gemm(h, H, convW16 + (size_t)l * H * H, H, nullptr, buf1, H, N, H, H, 0);
    hipLaunchKernelGGL(k_zero, dim3(4096), dim3(TB), 0, stream, buf2, NH);
    hipLaunchKernelGGL(k_edge_scatter, dim3(ceil_div((size_t)E * (H >> 2), TB)), dim3(TB),
                       0, stream, src, dst, dinv, buf1, buf2, E, H);
    hipLaunchKernelGGL(k_self_bias, dim3(ceil_div(NH, TB)), dim3(TB), 0, stream,
                       buf1, dinv, convB + (size_t)l * H, buf2, N, H);
    colstats(buf2, N, H);
    hipLaunchKernelGGL(k_bn_relu_res, dim3(ceil_div(NH, TB)), dim3(TB), 0, stream,
                       h, buf2, N, H, meanv, varv,
                       convG + (size_t)l * H, convBe + (size_t)l * H);
  }

  // 4) attention gate
  gemm(h, H, gW1_16, Hg, gB1, buf1, Hg, N, H, Hg, 1);
  hipLaunchKernelGGL(k_gate2, dim3(ceil_div((size_t)N, 8)), dim3(TB), 0, stream,
                     buf1, gW2, gB2, gate, N, Hg);

  // 5) segment softmax
  hipLaunchKernelGGL(k_fill, dim3(64), dim3(TB), 0, stream, gmax, -FLT_MAX, (size_t)B);
  hipLaunchKernelGGL(k_segmax, dim3(ceil_div(N, TB)), dim3(TB), 0, stream, gate, batch, gmax, N);
  hipLaunchKernelGGL(k_zero, dim3(64), dim3(TB), 0, stream, denom, (size_t)B);
  hipLaunchKernelGGL(k_expsum, dim3(ceil_div(N, TB)), dim3(TB), 0, stream,
                     gate, batch, gmax, eArr, denom, N);

  // 6) t = relu(h @ tW + tB); pooled = segsum(t*e)/denom
  gemm(h, H, tW16, H, tB, buf2, H, N, H, H, 1);
  hipLaunchKernelGGL(k_zero, dim3(1024), dim3(TB), 0, stream, pooled, (size_t)B * H);
  hipLaunchKernelGGL(k_pool, dim3(ceil_div((size_t)N * (H >> 2), TB)), dim3(TB), 0, stream,
                     buf2, eArr, batch, pooled, N, H);
  hipLaunchKernelGGL(k_pool_div, dim3(ceil_div((size_t)B * H, TB)), dim3(TB), 0, stream,
                     pooled, denom, B, H);

  // 7) MLP head
  gemm(pooled, H, mW1_16, H2, mB1, buf1, H2, B, H, H2, 0);
  colstats(buf1, B, H2);
  hipLaunchKernelGGL(k_bn_relu, dim3(ceil_div((size_t)B * H2, TB)), dim3(TB), 0, stream,
                     buf1, B, H2, meanv, varv, mG1, mBe1);

  gemm(buf1, H2, mW2_16, H2, mB2, buf2, H2, B, H2, H2, 0);
  colstats(buf2, B, H2);
  hipLaunchKernelGGL(k_bn_relu, dim3(ceil_div((size_t)B * H2, TB)), dim3(TB), 0, stream,
                     buf2, B, H2, meanv, varv, mG2, mBe2);

  float* out = (float*)d_out;
  gemm(buf2, H2, mW3_16, S, mB3, out, S, B, H2, S, 0);
  hipLaunchKernelGGL(k_sigmoid, dim3(ceil_div((size_t)B * S, TB)), dim3(TB), 0, stream,
                     out, (size_t)B * S);
}
